// PAM_89558658056622
// MI455X (gfx1250) — compile-verified, matrix operand order checked by reading, not run on hardware
//
#include <hip/hip_runtime.h>
#include <stdint.h>

// ---------------------------------------------------------------------------
// PAM (position attention): out = X @ softmax_rows(X^T X)  per batch
//   x   : [B, C, N] fp32   (B=8, C=256, N=48*48=2304)
//   out : [B, C, N] fp32
// Strategy (all matmuls on v_wmma_f32_16x16x32_bf16, fp32 accumulate):
//   K1: fp32 -> bf16 convert:  Xb [B,C,N]  and  Vb = transpose [B,N,C]
//   K2: row stats of S = Vb Vb^T :  M_n = rowmax, Zinv_n = 1/rowsumexp
//   K3: out[:, m-tile] = sum_n Xb[:,n] * exp(S[n,m]-M_n)*Zinv_n  (S recomputed)
// ---------------------------------------------------------------------------

#define NB 8
#define NC 256
#define NN 2304

typedef __attribute__((ext_vector_type(16))) __bf16        v16bf;
typedef __attribute__((ext_vector_type(8)))  float         v8f;
typedef __attribute__((ext_vector_type(4)))  unsigned int  u32x4;

union Frag16 { v16bf h; u32x4 q[2]; };

__device__ __forceinline__ unsigned short f32_to_bf16_rne(float f) {
  unsigned u = __float_as_uint(f);
  u += 0x7FFFu + ((u >> 16) & 1u);          // round-to-nearest-even
  return (unsigned short)(u >> 16);
}

__device__ __forceinline__ u32x4 ld_g16(const unsigned short* p) {
  return *(const u32x4*)p;                   // 16B -> global_load_b128
}

// ---------------------------------------------------------------------------
// Kernel 1: convert + transpose into bf16 operand layouts
// ---------------------------------------------------------------------------
__global__ __launch_bounds__(256)
void pam_prep(const float* __restrict__ x,
              unsigned short* __restrict__ Xb,
              unsigned short* __restrict__ Vb) {
  size_t idx = (size_t)blockIdx.x * blockDim.x + threadIdx.x;
  const size_t total = (size_t)NB * NC * NN;
  if (idx >= total) return;
  int n = (int)(idx % NN);
  int c = (int)((idx / NN) % NC);
  int b = (int)(idx / ((size_t)NN * NC));
  unsigned short h = f32_to_bf16_rne(x[idx]);
  Xb[idx] = h;                                   // [b, c, n]
  Vb[((size_t)b * NN + n) * NC + c] = h;         // [b, n, c]
}

// ---------------------------------------------------------------------------
// Kernel 2: per-row softmax stats of S = Vb Vb^T.
// One wave (32 threads) owns 16 rows; A-fragments (16x256) held in VGPRs.
// Two independent WMMA accumulator chains (even/odd K) avoid D->C RAW nops.
// Per-lane online (max, sumexp) state with a single exp per element.
// ---------------------------------------------------------------------------
__global__ __launch_bounds__(32)
void pam_stats(const unsigned short* __restrict__ Vb,
               float* __restrict__ Mrow,
               float* __restrict__ Zinv) {
  const int b    = blockIdx.y;
  const int n0   = blockIdx.x * 16;
  const int lane = threadIdx.x;
  const int col  = lane & 15;                    // M (for A) / N (for B) index
  const int kh   = (lane >> 4) * 8;              // K half select per ISA layout

  const unsigned short* arow = Vb + ((size_t)b * NN + n0 + col) * NC;
  Frag16 a[8];
#pragma unroll
  for (int kc = 0; kc < 8; ++kc) {               // K = 256 -> 8 x 32
    a[kc].q[0] = ld_g16(arow + kc * 32 + kh);
    a[kc].q[1] = ld_g16(arow + kc * 32 + kh + 16);
  }

  float mst[8], zst[8];
#pragma unroll
  for (int v = 0; v < 8; ++v) { mst[v] = -3.4e38f; zst[v] = 0.f; }

  for (int j0 = 0; j0 < NN; j0 += 16) {
    const unsigned short* brow = Vb + ((size_t)b * NN + j0 + col) * NC;
    // prefetch next tile's B row toward the WGP while this tile computes
    if (j0 + 16 < NN) __builtin_prefetch(brow + (size_t)16 * NC, 0, 1);

    v8f acca = {}, accb = {};
#pragma unroll
    for (int kc = 0; kc < 8; kc += 2) {
      Frag16 b0, b1;
      b0.q[0] = ld_g16(brow + kc * 32 + kh);
      b0.q[1] = ld_g16(brow + kc * 32 + kh + 16);
      b1.q[0] = ld_g16(brow + kc * 32 + 32 + kh);
      b1.q[1] = ld_g16(brow + kc * 32 + 32 + kh + 16);
      acca = __builtin_amdgcn_wmma_f32_16x16x32_bf16(
                 false, a[kc].h,     false, b0.h, (short)0, acca, false, false);
      accb = __builtin_amdgcn_wmma_f32_16x16x32_bf16(
                 false, a[kc + 1].h, false, b1.h, (short)0, accb, false, false);
    }
    v8f acc = acca + accb;

#pragma unroll
    for (int v = 0; v < 8; ++v) {                // online max / sumexp, 1 exp
      float s  = acc[v];
      float mo = mst[v];
      float mn = fmaxf(mo, s);
      float e1 = __expf(fminf(mo, s) - mn);      // exp(min-max); other factor==1
      zst[v] = (s > mo) ? (zst[v] * e1 + 1.0f) : (zst[v] + e1);
      mst[v] = mn;
    }
  }

  __shared__ float ml[256];
  __shared__ float zl[256];
#pragma unroll
  for (int v = 0; v < 8; ++v) {
    int row = v + ((lane >> 4) << 3);            // C/D layout: M = v + 8*(lane>=16)
    ml[row * 16 + col] = mst[v];
    zl[row * 16 + col] = zst[v];
  }
  __syncthreads();
  if (lane < 16) {
    float mm = -3.4e38f;
#pragma unroll
    for (int i = 0; i < 16; ++i) mm = fmaxf(mm, ml[lane * 16 + i]);
    float zz = 0.f;
#pragma unroll
    for (int i = 0; i < 16; ++i) zz += zl[lane * 16 + i] * __expf(ml[lane * 16 + i] - mm);
    Mrow[(size_t)b * NN + n0 + lane] = mm;
    Zinv[(size_t)b * NN + n0 + lane] = 1.0f / zz;
  }
}

// ---------------------------------------------------------------------------
// Kernel 3: out[:, m0:m0+16] for one batch. 8 waves / 256 threads.
// Per 32-row n-chunk: phase A computes S chunk [32x16] (16 WMMAs split
// 2-per-wave over K slices, LDS reduce), softmax weights packed bf16 to LDS,
// phase B: each wave accumulates its 32x16 output slab (2 WMMAs, K=32).
// ---------------------------------------------------------------------------
__global__ __launch_bounds__(256)
void pam_out(const unsigned short* __restrict__ Xb,
             const unsigned short* __restrict__ Vb,
             const float* __restrict__ Mrow,
             const float* __restrict__ Zinv,
             float* __restrict__ out) {
  const int b    = blockIdx.y;
  const int m0   = blockIdx.x * 16;
  const int tid  = threadIdx.x;
  const int lane = tid & 31;
  const int wave = tid >> 5;
  const int col  = lane & 15;
  const int kh   = (lane >> 4) * 8;

  __shared__ float          part[8 * 256];       // 8 partial 16x16 fp32 tiles
  __shared__ unsigned short pl[16 * 32];         // P chunk, B-fragment layout

  const int s  = wave & 1;                       // n-subtile (0/1)
  const int kp = wave >> 1;                      // K slice (0..3), 64 ch each
  const int c0 = wave * 32;                      // this wave's output rows

  const unsigned short* browm = Vb + ((size_t)b * NN + m0 + col) * NC;

  v8f acc0 = {}, acc1 = {};

  for (int n0 = 0; n0 < NN; n0 += 32) {
    // ---- phase A: partial S[n-subtile, m-tile] over 64-channel K slice ----
    const unsigned short* arow = Vb + ((size_t)b * NN + n0 + 16 * s + col) * NC;
    if (n0 + 32 < NN)                            // prefetch next chunk's A row
      __builtin_prefetch(arow + (size_t)32 * NC, 0, 1);

    v8f sp0 = {}, sp1 = {};
    {
      int k0 = kp * 64, k1 = kp * 64 + 32;
      Frag16 a0, b0, a1, b1;
      a0.q[0] = ld_g16(arow  + k0 + kh);  a0.q[1] = ld_g16(arow  + k0 + kh + 16);
      b0.q[0] = ld_g16(browm + k0 + kh);  b0.q[1] = ld_g16(browm + k0 + kh + 16);
      a1.q[0] = ld_g16(arow  + k1 + kh);  a1.q[1] = ld_g16(arow  + k1 + kh + 16);
      b1.q[0] = ld_g16(browm + k1 + kh);  b1.q[1] = ld_g16(browm + k1 + kh + 16);
      sp0 = __builtin_amdgcn_wmma_f32_16x16x32_bf16(
                false, a0.h, false, b0.h, (short)0, sp0, false, false);
      sp1 = __builtin_amdgcn_wmma_f32_16x16x32_bf16(
                false, a1.h, false, b1.h, (short)0, sp1, false, false);
    }
    v8f sp = sp0 + sp1;
#pragma unroll
    for (int v = 0; v < 8; ++v) {
      int row = v + ((lane >> 4) << 3);
      part[wave * 256 + row * 16 + col] = sp[v];
    }
    __syncthreads();

    // ---- reduce K-slice partials, apply softmax weights, pack bf16 ----
    for (int e = tid; e < 512; e += 256) {       // 512 elements of P[32x16]
      int k  = e & 31;
      int j  = e >> 5;
      int ss = k >> 4;                           // which n-subtile
      int r  = k & 15;
      float sum = part[(ss    ) * 256 + r * 16 + j]
                + part[(ss + 2) * 256 + r * 16 + j]
                + part[(ss + 4) * 256 + r * 16 + j]
                + part[(ss + 6) * 256 + r * 16 + j];
      int n = n0 + k;
      float p = __expf(sum - Mrow[(size_t)b * NN + n]) * Zinv[(size_t)b * NN + n];
      pl[j * 32 + k] = f32_to_bf16_rne(p);       // column-major: K contiguous
    }
    __syncthreads();

    // ---- phase B: acc += Xb[c-rows, n-chunk] * P ----
    Frag16 pb, x0, x1;
    pb.q[0] = *(const u32x4*)(pl + col * 32 + kh);
    pb.q[1] = *(const u32x4*)(pl + col * 32 + kh + 16);
    const unsigned short* xr0 = Xb + ((size_t)b * NC + c0 + col) * NN + n0;
    const unsigned short* xr1 = xr0 + (size_t)16 * NN;
    if (n0 + 32 < NN) {
      __builtin_prefetch(xr0 + 32, 0, 1);
      __builtin_prefetch(xr1 + 32, 0, 1);
    }
    x0.q[0] = ld_g16(xr0 + kh);  x0.q[1] = ld_g16(xr0 + kh + 16);
    x1.q[0] = ld_g16(xr1 + kh);  x1.q[1] = ld_g16(xr1 + kh + 16);
    acc0 = __builtin_amdgcn_wmma_f32_16x16x32_bf16(
               false, x0.h, false, pb.h, (short)0, acc0, false, false);
    acc1 = __builtin_amdgcn_wmma_f32_16x16x32_bf16(
               false, x1.h, false, pb.h, (short)0, acc1, false, false);
    __syncthreads();                             // protect part/pl reuse
  }

#pragma unroll
  for (int v = 0; v < 8; ++v) {
    int r = v + ((lane >> 4) << 3);
    out[((size_t)b * NC + c0 + r)      * NN + m0 + col] = acc0[v];
    out[((size_t)b * NC + c0 + 16 + r) * NN + m0 + col] = acc1[v];
  }
}

// ---------------------------------------------------------------------------
extern "C" void kernel_launch(void* const* d_in, const int* in_sizes, int n_in,
                              void* d_out, int out_size, void* d_ws, size_t ws_size,
                              hipStream_t stream) {
  (void)in_sizes; (void)n_in; (void)out_size; (void)ws_size;

  const float* x = (const float*)d_in[0];
  float* out = (float*)d_out;
  char* ws = (char*)d_ws;

  const size_t xb_bytes = (size_t)NB * NC * NN * sizeof(unsigned short); // 9.4 MB
  unsigned short* Xb = (unsigned short*)ws;
  unsigned short* Vb = (unsigned short*)(ws + xb_bytes);
  float* Mr = (float*)(ws + 2 * xb_bytes);
  float* Zi = Mr + (size_t)NB * NN;

  {
    const size_t total = (size_t)NB * NC * NN;
    const int blocks = (int)((total + 255) / 256);
    pam_prep<<<blocks, 256, 0, stream>>>(x, Xb, Vb);
  }
  pam_stats<<<dim3(NN / 16, NB), 32, 0, stream>>>(Vb, Mr, Zi);
  pam_out<<<dim3(NN / 16, NB), 256, 0, stream>>>(Xb, Vb, Mr, Zi, out);
}
